// DecodlerLayer_63187558859181
// MI455X (gfx1250) — matrix operand order repaired and checked by reading, not compiled, not run on hardware
//
#include <hip/hip_runtime.h>

// ---------------------------------------------------------------------------
// MI455X (gfx1250) decoder layer: bf16 WMMA GEMMs + column softmax.
// B=4, S=2048, PAST=2048, D_IN=D_H=2048, D_FF=8192, L=PAST+S=4096.
// Everything routed through V_WMMA_F32_16X16X32_BF16 (fp32 accumulate).
// Block tile 128x128 (8 waves), wave tile 32x64 = 2x4 WMMA accumulators.
// ---------------------------------------------------------------------------

typedef __attribute__((ext_vector_type(16))) __bf16 v16bf;
typedef __attribute__((ext_vector_type(8)))  float  v8f;

#define BM 128
#define BN 128
#define BK 32
#define LDA_S 40   // LDS row stride in ushorts (80B: 16B-aligned, bank-spread)
#define LDB_S 40

__device__ __forceinline__ unsigned short f32_to_bf16(float f) {
  unsigned int u = __float_as_uint(f);
  unsigned int r = (u + 0x7FFFu + ((u >> 16) & 1u)) >> 16;
  return (unsigned short)r;
}

// ---------------------------------------------------------------------------
// Generic batched GEMM: C[M,N] = A[M,K] * Bt[N,K]^T (+ bias[n]).
// A, Bt are bf16 (as ushort). Optional f32 output Cf and bf16 output Ch
// (optionally transposed), each with segment remap:
//   eff = blockIdx.z + m/segS ; row = m%segS + rowoff
//   Cf[eff*cf_stride + row*cf_ld + n]
//   Ch[eff*ch_stride + (transposed ? n*ch_ld + row : row*ch_ld + n)]
// ---------------------------------------------------------------------------
__global__ __launch_bounds__(256) void gemm_bf16_kernel(
    const unsigned short* __restrict__ A, long lda, long a_bstride,
    const unsigned short* __restrict__ Bt, long ldb, long b_bstride,
    const float* __restrict__ bias,
    float* __restrict__ Cf, long cf_ld, long cf_stride, int cf_rowoff,
    unsigned short* __restrict__ Ch, long ch_ld, long ch_stride, int ch_rowoff,
    int ch_transposed, int segS, int M, int N, int K)
{
  __shared__ __align__(16) unsigned short As[BM * LDA_S];
  __shared__ __align__(16) unsigned short Bs[BN * LDB_S];

  const int bz = blockIdx.z;
  const int n0 = blockIdx.x * BN;
  const int m0 = blockIdx.y * BM;
  A  += (long)bz * a_bstride;
  Bt += (long)bz * b_bstride;

  const int tid   = threadIdx.x;
  const int lane  = tid & 31;
  const int wid   = tid >> 5;    // 0..7
  const int wm    = wid >> 1;    // 0..3 : 32-row slab
  const int wn    = wid & 1;     // 0..1 : 64-col slab
  const int lhalf = lane >> 4;   // 0/1
  const int l15   = lane & 15;

  v8f acc[2][4];
  #pragma unroll
  for (int i = 0; i < 2; i++)
    #pragma unroll
    for (int j = 0; j < 4; j++)
      #pragma unroll
      for (int e = 0; e < 8; e++) acc[i][j][e] = 0.0f;

  for (int k0 = 0; k0 < K; k0 += BK) {
    // ---- global loads (16B chunks; all dims multiples of tile sizes) ----
    // A tile: 128 rows x 32 k = 512 chunks of 8 bf16; 2 chunks/thread.
    // B tile: 128 rows x 32 k = 512 chunks;            2 chunks/thread.
    uint4 a_reg[2], b_reg[2];
    #pragma unroll
    for (int c = 0; c < 2; c++) {
      int chunk = tid + c * 256;
      int row = chunk >> 2;
      int col = (chunk & 3) << 3;
      a_reg[c] = *(const uint4*)(A  + (long)(m0 + row) * lda + k0 + col);
      b_reg[c] = *(const uint4*)(Bt + (long)(n0 + row) * ldb + k0 + col);
    }
    __syncthreads();                      // previous compute done with LDS
    #pragma unroll
    for (int c = 0; c < 2; c++) {
      int chunk = tid + c * 256;
      int row = chunk >> 2;
      int col = (chunk & 3) << 3;
      *(uint4*)(As + row * LDA_S + col) = a_reg[c];
      *(uint4*)(Bs + row * LDB_S + col) = b_reg[c];
    }
    __syncthreads();

    // ---- fragments (CDNA5 wave32 WMMA VGPR layouts) ----
    v16bf afrag[2], bfrag[4];
    #pragma unroll
    for (int i = 0; i < 2; i++) {
      // A: lane row = l15, elems 0..7 -> K = lhalf*8 + e, elems 8..15 -> +16
      const unsigned short* p = As + (wm * 32 + i * 16 + l15) * LDA_S + lhalf * 8;
      ((uint4*)&afrag[i])[0] = *(const uint4*)(p);
      ((uint4*)&afrag[i])[1] = *(const uint4*)(p + 16);
    }
    #pragma unroll
    for (int j = 0; j < 4; j++) {
      // B: lane col = l15, elems 0..7 -> K = lhalf*8 + e, elems 8..15 -> +16 (Bt is [N][K])
      const unsigned short* p = Bs + (wn * 64 + j * 16 + l15) * LDB_S + lhalf * 8;
      ((uint4*)&bfrag[j])[0] = *(const uint4*)(p);
      ((uint4*)&bfrag[j])[1] = *(const uint4*)(p + 16);
    }
    #pragma unroll
    for (int i = 0; i < 2; i++)
      #pragma unroll
      for (int j = 0; j < 4; j++)
        acc[i][j] = __builtin_amdgcn_wmma_f32_16x16x32_bf16(
            false, afrag[i], false, bfrag[j], (short)0, acc[i][j], false, false);
  }

  // ---- epilogue: C row = r + lhalf*8 (per vgpr), col = l15 ----
  #pragma unroll
  for (int i = 0; i < 2; i++) {
    #pragma unroll
    for (int j = 0; j < 4; j++) {
      int n = n0 + wn * 64 + j * 16 + l15;
      float bv = bias ? bias[n] : 0.0f;
      int mrow0 = m0 + wm * 32 + i * 16 + lhalf * 8;
      #pragma unroll
      for (int r = 0; r < 8; r++) {
        int m = mrow0 + r;
        float val = acc[i][j][r] + bv;
        long eff = (long)bz + (long)(m / segS);
        long rr  = (long)(m % segS);
        if (Cf)
          Cf[eff * cf_stride + (rr + cf_rowoff) * cf_ld + n] = val;
        if (Ch) {
          unsigned short h = f32_to_bf16(val);
          if (ch_transposed)
            Ch[eff * ch_stride + (long)n * ch_ld + (rr + ch_rowoff)] = h;
          else
            Ch[eff * ch_stride + (rr + ch_rowoff) * ch_ld + n] = h;
        }
      }
    }
  }
}

// f32 -> bf16 straight convert
__global__ void cvt_bf16_kernel(const float* __restrict__ src,
                                unsigned short* __restrict__ dst, long n) {
  long i = (long)blockIdx.x * blockDim.x + threadIdx.x;
  long stride = (long)gridDim.x * blockDim.x;
  for (; i < n; i += stride) dst[i] = f32_to_bf16(src[i]);
}

// batched f32 [R,C] -> bf16 [C, dld] transpose-convert via LDS tiles
__global__ void transpose_cvt_kernel(const float* __restrict__ src, long s_bstride,
                                     int R, int C,
                                     unsigned short* __restrict__ dst, long d_bstride,
                                     int dld) {
  __shared__ float t[32][33];
  const float* s = src + (long)blockIdx.z * s_bstride;
  unsigned short* d = dst + (long)blockIdx.z * d_bstride;
  int c0 = blockIdx.x * 32, r0 = blockIdx.y * 32;
  int tx = threadIdx.x, ty = threadIdx.y;  // 32 x 8
  #pragma unroll
  for (int i = 0; i < 4; i++) {
    int r = r0 + ty + i * 8;
    t[ty + i * 8][tx] = s[(long)r * C + c0 + tx];
  }
  __syncthreads();
  #pragma unroll
  for (int i = 0; i < 4; i++) {
    int c = c0 + ty + i * 8;
    d[(long)c * dld + r0 + tx] = f32_to_bf16(t[tx][ty + i * 8]);
  }
}

// copy cache f32 into concatenated output region (+ optional bf16 shadow)
__global__ void cache_copy_kernel(const float* __restrict__ src,
                                  float* __restrict__ dstF,
                                  unsigned short* __restrict__ dstH,
                                  long per_batch, long out_bstride, long total) {
  long i = (long)blockIdx.x * blockDim.x + threadIdx.x;
  long stride = (long)gridDim.x * blockDim.x;
  for (; i < total; i += stride) {
    long b = i / per_batch, r = i % per_batch;
    float v = src[i];
    long o = b * out_bstride + r;
    dstF[o] = v;
    if (dstH) dstH[o] = f32_to_bf16(v);
  }
}

// softmax over the QUERY axis (dim=1 of [B,S,L]); one thread per (b, key col)
__global__ void softmax_q_kernel(const float* __restrict__ scores,
                                 unsigned short* __restrict__ attn,
                                 int S, int L, long bstride) {
  int l = blockIdx.x * blockDim.x + threadIdx.x;
  int b = blockIdx.y;
  const float* s = scores + (long)b * bstride + l;
  unsigned short* a = attn + (long)b * bstride + l;
  float mx = -3.402823466e38f;
  for (int q = 0; q < S; q++) mx = fmaxf(mx, s[(long)q * L]);
  float sum = 0.0f;
  for (int q = 0; q < S; q++) sum += __expf(s[(long)q * L] - mx);
  float inv = 1.0f / sum;
  for (int q = 0; q < S; q++)
    a[(long)q * L] = f32_to_bf16(__expf(s[(long)q * L] - mx) * inv);
}

// ---------------------------------------------------------------------------
extern "C" void kernel_launch(void* const* d_in, const int* in_sizes, int n_in,
                              void* d_out, int out_size, void* d_ws, size_t ws_size,
                              hipStream_t stream) {
  (void)in_sizes; (void)n_in; (void)out_size; (void)ws_size;

  const int B = 4, S = 2048, PAST = 2048, D = 2048, DFF = 8192, L = PAST + S;
  const long BS = (long)B * S;           // 8192
  const long SD = (long)S * D;           // 4,194,304
  const long LD = (long)L * D;           // 8,388,608
  const long SL = (long)S * L;           // 8,388,608
  const long DL = (long)D * L;           // 8,388,608

  const float* x    = (const float*)d_in[0];
  const float* kc   = (const float*)d_in[1];
  const float* vc   = (const float*)d_in[2];
  const float* Wq   = (const float*)d_in[3];  const float* bq   = (const float*)d_in[4];
  const float* Wk   = (const float*)d_in[5];  const float* bk   = (const float*)d_in[6];
  const float* Wv   = (const float*)d_in[7];  const float* bv   = (const float*)d_in[8];
  const float* Wff  = (const float*)d_in[9];  const float* bff  = (const float*)d_in[10];
  const float* Wout = (const float*)d_in[11]; const float* bout = (const float*)d_in[12];

  float* outp = (float*)d_out;                 // [B,S,D]
  float* kout = outp + BS * D;                 // [B,L,D]
  float* vout = kout + (long)B * LD;           // [B,L,D]

  // workspace layout (bytes), aggressively reused
  char* ws = (char*)d_ws;
  unsigned short* xh     = (unsigned short*)(ws + 0);           // bf16 x  [8192,2048]
  unsigned short* wq_t   = (unsigned short*)(ws + 33554432LL);  // Wq^T
  unsigned short* wk_t   = wq_t + (long)D * D;
  unsigned short* wv_t   = wk_t + (long)D * D;
  unsigned short* kh     = (unsigned short*)(ws + 134217728LL); // bf16 k  [B,L,D]
  unsigned short* vt     = (unsigned short*)(ws + 201326592LL); // bf16 v^T [B,D,L]
  unsigned short* qh     = (unsigned short*)(ws + 268435456LL); // bf16 q  [B,S,D]
  float*          scores = (float*)(ws + 0);                    // f32 [B,S,L] (reuses xh/W*)
  unsigned short* attn   = (unsigned short*)(ws + 301989888LL); // bf16 [B,S,L]
  unsigned short* ctxh   = (unsigned short*)(ws + 268435456LL); // bf16 [B,S,D] (reuses qh)
  unsigned short* hh     = (unsigned short*)(ws + 0);           // bf16 [8192,8192] (reuses scores)
  unsigned short* wff_t  = (unsigned short*)(ws + 301989888LL); // Wff^T (reuses attn)
  unsigned short* wout_t = (unsigned short*)(ws + 335544320LL); // Wout^T

  // ---- phase 1: converts / transposes / cache copies ----
  cvt_bf16_kernel<<<2048, 256, 0, stream>>>(x, xh, BS * D);
  transpose_cvt_kernel<<<dim3(D/32, D/32, 1), dim3(32, 8), 0, stream>>>(Wq, 0, D, D, wq_t, 0, D);
  transpose_cvt_kernel<<<dim3(D/32, D/32, 1), dim3(32, 8), 0, stream>>>(Wk, 0, D, D, wk_t, 0, D);
  transpose_cvt_kernel<<<dim3(D/32, D/32, 1), dim3(32, 8), 0, stream>>>(Wv, 0, D, D, wv_t, 0, D);
  cache_copy_kernel<<<2048, 256, 0, stream>>>(kc, kout, kh, (long)PAST * D, LD, (long)B * PAST * D);
  cache_copy_kernel<<<2048, 256, 0, stream>>>(vc, vout, (unsigned short*)nullptr,
                                              (long)PAST * D, LD, (long)B * PAST * D);
  transpose_cvt_kernel<<<dim3(D/32, PAST/32, B), dim3(32, 8), 0, stream>>>(
      vc, (long)PAST * D, PAST, D, vt, DL, L);

  // ---- phase 2: projections  M=8192, N=2048, K=2048 ----
  dim3 gproj(D / BN, BS / BM, 1);
  // q -> qh (bf16)
  gemm_bf16_kernel<<<gproj, 256, 0, stream>>>(
      xh, D, 0, wq_t, D, 0, bq,
      nullptr, 0, 0, 0,
      qh, D, 0, 0, 0, (int)BS, (int)BS, D, D);
  // k -> kout (f32, rows PAST..) and kh (bf16, rows PAST..)
  gemm_bf16_kernel<<<gproj, 256, 0, stream>>>(
      xh, D, 0, wk_t, D, 0, bk,
      kout, D, LD, PAST,
      kh, D, LD, PAST, 0, S, (int)BS, D, D);
  // v -> vout (f32, rows PAST..) and vt (bf16 transposed: [b][d][PAST+s])
  gemm_bf16_kernel<<<gproj, 256, 0, stream>>>(
      xh, D, 0, wv_t, D, 0, bv,
      vout, D, LD, PAST,
      vt, L, DL, PAST, 1, S, (int)BS, D, D);

  // ---- phase 3: scores = q @ k^T  (batched; M=S, N=L, K=D), then softmax ----
  gemm_bf16_kernel<<<dim3(L / BN, S / BM, B), 256, 0, stream>>>(
      qh, D, SD, kh, D, LD, nullptr,
      scores, L, SL, 0,
      nullptr, 0, 0, 0, 0, S, S, L, D);
  softmax_q_kernel<<<dim3(L / 256, B), 256, 0, stream>>>(scores, attn, S, L, SL);

  // ---- phase 4: ctx = attn @ v  (batched; M=S, N=D, K=L) ----
  gemm_bf16_kernel<<<dim3(D / BN, S / BM, B), 256, 0, stream>>>(
      attn, L, SL, vt, L, DL, nullptr,
      nullptr, 0, 0, 0,
      ctxh, D, SD, 0, 0, S, S, D, L);

  // ---- phase 5: h = ctx @ Wff + bff  (M=8192, N=8192, K=2048) ----
  transpose_cvt_kernel<<<dim3(DFF/32, D/32, 1), dim3(32, 8), 0, stream>>>(Wff, 0, D, DFF, wff_t, 0, D);
  gemm_bf16_kernel<<<dim3(DFF / BN, BS / BM, 1), 256, 0, stream>>>(
      ctxh, D, 0, wff_t, D, 0, bff,
      nullptr, 0, 0, 0,
      hh, DFF, 0, 0, 0, (int)BS, (int)BS, DFF, D);

  // ---- phase 6: out = h @ Wout + bout  (M=8192, N=2048, K=8192) ----
  transpose_cvt_kernel<<<dim3(D/32, DFF/32, 1), dim3(32, 8), 0, stream>>>(Wout, 0, DFF, D, wout_t, 0, DFF);
  gemm_bf16_kernel<<<dim3(D / BN, BS / BM, 1), 256, 0, stream>>>(
      hh, DFF, 0, wout_t, DFF, 0, bout,
      outp, D, 0, 0,
      nullptr, 0, 0, 0, 0, (int)BS, (int)BS, D, DFF);
}
